// SparseInputPerceiver_24077586661751
// MI455X (gfx1250) — matrix operand order repaired and checked by reading, not compiled, not run on hardware
//
#include <hip/hip_runtime.h>

// ---------------------------------------------------------------------------
// Types / helpers
// ---------------------------------------------------------------------------
typedef __bf16 bf16_t;
typedef __attribute__((ext_vector_type(16))) __bf16 v16bf;
typedef __attribute__((ext_vector_type(8)))  __bf16 v8bf;
typedef __attribute__((ext_vector_type(8)))  float  v8f;

#define DEV __device__ __forceinline__

DEV float bf2f(bf16_t x) {
    unsigned short u = __builtin_bit_cast(unsigned short, x);
    unsigned int   v = ((unsigned int)u) << 16;
    return __builtin_bit_cast(float, v);
}
DEV bf16_t f2bf(float f) {
    unsigned int u = __builtin_bit_cast(unsigned int, f);
    u += 0x7FFFu + ((u >> 16) & 1u);   // round-to-nearest-even
    return __builtin_bit_cast(bf16_t, (unsigned short)(u >> 16));
}

// WMMA bf16 16x16x32, f32 accumulate  (codegen-confirmed builtin)
DEV v8f wmma_bf16(v16bf a, v16bf b, v8f c) {
    return __builtin_amdgcn_wmma_f32_16x16x32_bf16(false, a, false, b,
                                                   (short)0, c, false, false);
}

// A-operand (16x32 bf16, row-major source): lane<16 -> row=lane, K=0..15;
// lane>=16 -> row=lane-16, K=16..31
DEV v16bf wmma_load_a(const bf16_t* A, int lda, int row0, int kbase, int lane) {
    int r  = row0 + (lane & 15);
    int k0 = kbase + ((lane & 16) ? 16 : 0);
    const bf16_t* p = A + (size_t)r * lda + k0;
    v16bf out;
    ((v8bf*)&out)[0] = *(const v8bf*)(p);
    ((v8bf*)&out)[1] = *(const v8bf*)(p + 8);
    return out;
}
// B-operand (32x16 bf16) from transposed storage Bt[N][ldb] (K contiguous):
// lane<16 -> col=lane, K=0..15 ; lane>=16 -> col=lane-16, K=16..31
DEV v16bf wmma_load_b(const bf16_t* Bt, int ldb, int col0, int kbase, int lane) {
    int c  = col0 + (lane & 15);
    int k0 = kbase + ((lane & 16) ? 16 : 0);
    return *(const v16bf*)(Bt + (size_t)c * ldb + k0);
}

DEV v8f vzero8() { v8f z = {0.f,0.f,0.f,0.f,0.f,0.f,0.f,0.f}; return z; }

// ---------------------------------------------------------------------------
// Model constants
// ---------------------------------------------------------------------------
#define BQ   2
#define CCH  3
#define HH   48
#define WW   48
#define TT   8
#define MM   (HH * WW * TT)     /* 18432 */
#define DLAT 512
#define NLAT 128
#define HEADS 8
#define DHEAD 75
#define DPAD 96
#define DIN_PAD 416             /* 390 -> 416 */
#define INNER_PAD 608           /* 600 -> 608 */
#define NFFQ 64

// ---------------------------------------------------------------------------
// 1) Sparsify: stable partition (mask-true first) via per-batch block scan
// ---------------------------------------------------------------------------
DEV int calc_mask(const float* inp, int b, int g) {
    int t = g & 7;
    int x = (g >> 3) % WW;
    int y = g / (WW * TT);
    float m = -3.4e38f;
#pragma unroll
    for (int c = 0; c < CCH; ++c) {
        float v = inp[((((size_t)t * BQ + b) * CCH + c) * HH + y) * WW + x];
        m = fmaxf(m, v);
    }
    return (fabsf(m) > 0.0f) ? 1 : 0;
}

__global__ void k_sparsify(const float* __restrict__ inp,
                           int* __restrict__ order, float* __restrict__ maskf) {
    const int b = blockIdx.x;
    const int tid = threadIdx.x;
    __shared__ int sdata[1024];

    // pass 1: count trues
    int local = 0;
    for (int base = 0; base < MM; base += 1024)
        local += calc_mask(inp, b, base + tid);
    sdata[tid] = local;
    __syncthreads();
    for (int s = 512; s > 0; s >>= 1) {
        if (tid < s) sdata[tid] += sdata[tid + s];
        __syncthreads();
    }
    const int totalTrue = sdata[0];
    __syncthreads();

    // pass 2: chunked inclusive scan -> stable destinations
    int runT = 0;
    for (int base = 0; base < MM; base += 1024) {
        int g = base + tid;
        int m = calc_mask(inp, b, g);
        sdata[tid] = m;
        __syncthreads();
        for (int off = 1; off < 1024; off <<= 1) {
            int v = (tid >= off) ? sdata[tid - off] : 0;
            __syncthreads();
            sdata[tid] += v;
            __syncthreads();
        }
        int incl = sdata[tid];
        int truesBefore  = runT + incl - m;
        int falsesBefore = g - truesBefore;
        int dest = m ? truesBefore : (totalTrue + falsesBefore);
        order[(size_t)b * MM + dest] = g;
        maskf[(size_t)b * MM + dest] = m ? 1.0f : 0.0f;
        runT += sdata[1023];
        __syncthreads();
    }
}

// ---------------------------------------------------------------------------
// 2) Fourier encode -> bf16 data [B*MM][416]
// ---------------------------------------------------------------------------
__global__ void k_encode(const float* __restrict__ inp,
                         const int* __restrict__ order,
                         const float* __restrict__ maskf,
                         bf16_t* __restrict__ data) {
    int idx = blockIdx.x * blockDim.x + threadIdx.x;
    if (idx >= BQ * MM) return;
    int b = idx / MM;
    int g = order[idx];
    float mval = maskf[idx];

    int t = g & 7;
    int x = (g >> 3) % WW;
    int y = g / (WW * TT);

    bf16_t* row = data + (size_t)idx * DIN_PAD;
#pragma unroll
    for (int c = 0; c < CCH; ++c) {
        float v = inp[((((size_t)t * BQ + b) * CCH + c) * HH + y) * WW + x];
        row[c] = f2bf(v * mval);
    }
    float coords[3] = { (float)y * mval, (float)x * mval, (float)t * mval };
    int p = CCH;
    for (int j = 0; j < 3; ++j) {
        float cv = coords[j];
        float inv = 1.0f;
        for (int k = 0; k < NFFQ; ++k) {
            float xs = cv * inv;
            row[p + k]        = f2bf(__sinf(xs));
            row[p + NFFQ + k] = f2bf(__cosf(xs));
            inv *= 0.5f;
        }
        row[p + 2 * NFFQ] = f2bf(cv);
        p += 2 * NFFQ + 1;
    }
    for (int k = 390; k < DIN_PAD; ++k) row[k] = f2bf(0.0f);
}

// ---------------------------------------------------------------------------
// 3) Weight cast: src f32 [K][N] row-major -> dst bf16 [Npad][Kpad] (transposed)
// ---------------------------------------------------------------------------
__global__ void k_cast_wT(const float* __restrict__ src, int K, int N,
                          int Kpad, int Npad, bf16_t* __restrict__ dst) {
    int id = blockIdx.x * blockDim.x + threadIdx.x;
    if (id >= Npad * Kpad) return;
    int n = id / Kpad;
    int k = id % Kpad;
    float v = (k < K && n < N) ? src[(size_t)k * N + n] : 0.0f;
    dst[id] = f2bf(v);
}

__global__ void k_init_z(const float* __restrict__ lat,
                         const float* __restrict__ pos, float* __restrict__ zf) {
    int id = blockIdx.x * blockDim.x + threadIdx.x;
    if (id >= BQ * NLAT * DLAT) return;
    int i = id % (NLAT * DLAT);
    zf[id] = lat[i] + pos[i];
}

__global__ void k_f32_to_bf16(const float* __restrict__ s, bf16_t* __restrict__ d, int n) {
    int id = blockIdx.x * blockDim.x + threadIdx.x;
    if (id < n) d[id] = f2bf(s[id]);
}

// ---------------------------------------------------------------------------
// 4) QKV GEMM: block = 8 row-tiles x one 64-col group; B tile staged in LDS;
//    each wave reuses its A fragment for 4 WMMAs (16x64 tile).
//    mode 0: Q -> [b][h][n][96]   mode 1: K [b][h][m][96], Vt [b][h][80][Mld]
// ---------------------------------------------------------------------------
__global__ void __launch_bounds__(256)
k_gemm_qkv(const bf16_t* __restrict__ A, int lda,
           const bf16_t* __restrict__ Wt, int ldw, int ncol,
           int Klen, int perb,
           bf16_t* __restrict__ Kout, bf16_t* __restrict__ Vtout,
           bf16_t* __restrict__ Qout, int Mld, int mode) {
    const int lane = threadIdx.x & 31;
    const int warp = threadIdx.x >> 5;
    const int cg = blockIdx.x;              // 64-column group
    const int rt = blockIdx.y * 8 + warp;   // 16-row tile per wave

    __shared__ bf16_t Bs[64 * 32];          // [col][k] tile, 4KB

    v8f acc[4];
#pragma unroll
    for (int j = 0; j < 4; ++j) acc[j] = vzero8();

    const int sc_c  = threadIdx.x >> 2;          // 0..63  staging col
    const int sc_k  = (threadIdx.x & 3) * 8;     // 0,8,16,24
    const int wcol  = cg * 64 + sc_c;

    for (int k0 = 0; k0 < Klen; k0 += 32) {
        __syncthreads();
        v8bf wv;
#pragma unroll
        for (int q = 0; q < 8; ++q) wv[q] = f2bf(0.0f);
        if (wcol < ncol) wv = *(const v8bf*)(Wt + (size_t)wcol * ldw + k0 + sc_k);
        *(v8bf*)(&Bs[sc_c * 32 + sc_k]) = wv;
        __syncthreads();

        v16bf a = wmma_load_a(A, lda, rt * 16, k0, lane);
#pragma unroll
        for (int j = 0; j < 4; ++j) {
            v16bf b = wmma_load_b(Bs, 32, j * 16, 0, lane);
            acc[j] = wmma_bf16(a, b, acc[j]);
        }
    }

    const int rbase = rt * 16 + ((lane & 16) ? 8 : 0);
#pragma unroll
    for (int j = 0; j < 4; ++j) {
        int c = cg * 64 + j * 16 + (lane & 15);
        if (c >= ncol) continue;
#pragma unroll
        for (int i = 0; i < 8; ++i) {
            int r  = rbase + i;
            int bb = r / perb;
            int m  = r % perb;
            float v = acc[j][i];
            if (mode == 0) {
                if (c < 600) {
                    int h = c / DHEAD, d = c % DHEAD;
                    Qout[(((size_t)bb * HEADS + h) * NLAT + m) * DPAD + d] = f2bf(v);
                }
            } else {
                if (c < 600) {
                    int h = c / DHEAD, d = c % DHEAD;
                    Kout[(((size_t)bb * HEADS + h) * Mld + m) * DPAD + d] = f2bf(v);
                } else if (c < 1200) {
                    int cc = c - 600;
                    int h = cc / DHEAD, d = cc % DHEAD;
                    Vtout[(((size_t)bb * HEADS + h) * 80 + d) * (size_t)Mld + m] = f2bf(v);
                }
            }
        }
    }
}

// ---------------------------------------------------------------------------
// 5) Generic GEMM + bias, same tiling as above.
//    mode 0: Cf = A@Wt + bias                         (f32)
//    mode 1: Cb = gelu(A@Wt + bias)                   (bf16)
//    mode 2: Z  = (A@Wt+bias)*scale + 2*Z ; Zb = bf16(Z)   (ReZero, N==512)
// ---------------------------------------------------------------------------
__global__ void __launch_bounds__(256)
k_gemm_bias(const bf16_t* __restrict__ A, int lda,
            const bf16_t* __restrict__ Wt, int ldw, int N,
            int Klen, const float* __restrict__ bias,
            float* __restrict__ Cf, bf16_t* __restrict__ Cb,
            const float* __restrict__ scale, float* __restrict__ Z,
            bf16_t* __restrict__ Zb, int mode) {
    const int lane = threadIdx.x & 31;
    const int warp = threadIdx.x >> 5;
    const int cg = blockIdx.x;
    const int rt = blockIdx.y * 8 + warp;

    __shared__ bf16_t Bs[64 * 32];

    v8f acc[4];
#pragma unroll
    for (int j = 0; j < 4; ++j) acc[j] = vzero8();

    const int sc_c = threadIdx.x >> 2;
    const int sc_k = (threadIdx.x & 3) * 8;
    const int wcol = cg * 64 + sc_c;

    for (int k0 = 0; k0 < Klen; k0 += 32) {
        __syncthreads();
        v8bf wv;
#pragma unroll
        for (int q = 0; q < 8; ++q) wv[q] = f2bf(0.0f);
        if (wcol < N) wv = *(const v8bf*)(Wt + (size_t)wcol * ldw + k0 + sc_k);
        *(v8bf*)(&Bs[sc_c * 32 + sc_k]) = wv;
        __syncthreads();

        v16bf a = wmma_load_a(A, lda, rt * 16, k0, lane);
#pragma unroll
        for (int j = 0; j < 4; ++j) {
            v16bf b = wmma_load_b(Bs, 32, j * 16, 0, lane);
            acc[j] = wmma_bf16(a, b, acc[j]);
        }
    }

    const int rbase = rt * 16 + ((lane & 16) ? 8 : 0);
#pragma unroll
    for (int j = 0; j < 4; ++j) {
        int c = cg * 64 + j * 16 + (lane & 15);
        if (c >= N) continue;
        float bv = bias[c];
#pragma unroll
        for (int i = 0; i < 8; ++i) {
            int r = rbase + i;
            float v = acc[j][i] + bv;
            if (mode == 0) {
                Cf[(size_t)r * N + c] = v;
            } else if (mode == 1) {
                float g = 0.5f * v * (1.0f + erff(v * 0.70710678118654752f));
                Cb[(size_t)r * N + c] = f2bf(g);
            } else {
                size_t zi = (size_t)r * DLAT + c;
                float nv = v * (*scale) + 2.0f * Z[zi];
                Z[zi]  = nv;
                Zb[zi] = f2bf(nv);
            }
        }
    }
}

// ---------------------------------------------------------------------------
// 6) Flash attention: one wave per (b, h, 16-row latent tile).
//    Q [b][h][128][96], K [b][h][Mctx][96], Vt [b][h][80][Mctx]
//    O [b*128][608] bf16 (col = h*75 + d)
// ---------------------------------------------------------------------------
__global__ void __launch_bounds__(256)
k_attn(const bf16_t* __restrict__ Q, const bf16_t* __restrict__ Kt,
       const bf16_t* __restrict__ Vt, const float* __restrict__ maskf,
       int Mctx, bf16_t* __restrict__ O) {
    const int lane = threadIdx.x & 31;
    const int warp = threadIdx.x >> 5;
    int w  = blockIdx.x * 8 + warp;     // 128 waves total
    int b  = w >> 6;
    int h  = (w >> 3) & 7;
    int nt = w & 7;

    const size_t bh = (size_t)(b * HEADS + h);
    const bf16_t* Qb = Q + (bh * NLAT + nt * 16) * DPAD;
    const bf16_t* Kb = Kt + bh * (size_t)Mctx * DPAD;
    const bf16_t* Vb = Vt + bh * (size_t)80 * Mctx;
    const float*  mk = maskf ? (maskf + (size_t)b * Mctx) : nullptr;

    v16bf aq0 = wmma_load_a(Qb, DPAD, 0, 0, lane);
    v16bf aq1 = wmma_load_a(Qb, DPAD, 0, 32, lane);
    v16bf aq2 = wmma_load_a(Qb, DPAD, 0, 64, lane);

    v8f oacc[5];
    float rmax[8], rsum[8];
#pragma unroll
    for (int t = 0; t < 5; ++t) oacc[t] = vzero8();
#pragma unroll
    for (int i = 0; i < 8; ++i) { rmax[i] = -3.4e38f; rsum[i] = 0.0f; }

    __shared__ bf16_t ptile[8][16][32];
    const float sc = 0.11547005383792516f;   // 75^-0.5
    const int half8 = (lane & 16) ? 8 : 0;
    const int col   = lane & 15;

    for (int m0 = 0; m0 < Mctx; m0 += 32) {
        v8f s0 = vzero8(), s1 = vzero8();
        {
            v16bf b0 = wmma_load_b(Kb, DPAD, m0,      0, lane);
            v16bf b1 = wmma_load_b(Kb, DPAD, m0 + 16, 0, lane);
            s0 = wmma_bf16(aq0, b0, s0); s1 = wmma_bf16(aq0, b1, s1);
            b0 = wmma_load_b(Kb, DPAD, m0,      32, lane);
            b1 = wmma_load_b(Kb, DPAD, m0 + 16, 32, lane);
            s0 = wmma_bf16(aq1, b0, s0); s1 = wmma_bf16(aq1, b1, s1);
            b0 = wmma_load_b(Kb, DPAD, m0,      64, lane);
            b1 = wmma_load_b(Kb, DPAD, m0 + 16, 64, lane);
            s0 = wmma_bf16(aq2, b0, s0); s1 = wmma_bf16(aq2, b1, s1);
        }
        float mv0 = 1.f, mv1 = 1.f;
        if (mk) { mv0 = mk[m0 + col]; mv1 = mk[m0 + 16 + col]; }

        float p0[8], p1[8];
#pragma unroll
        for (int i = 0; i < 8; ++i) {
            float a0 = (mv0 > 0.f) ? s0[i] * sc : -3.0e38f;
            float a1 = (mv1 > 0.f) ? s1[i] * sc : -3.0e38f;
            float cm = fmaxf(a0, a1);
#pragma unroll
            for (int d = 1; d < 16; d <<= 1) cm = fmaxf(cm, __shfl_xor(cm, d, 32));
            float nm = fmaxf(rmax[i], cm);
            float alpha = __expf(rmax[i] - nm);
            float e0 = (a0 <= -1.0e37f) ? 0.f : __expf(a0 - nm);
            float e1 = (a1 <= -1.0e37f) ? 0.f : __expf(a1 - nm);
            float ps = e0 + e1;
#pragma unroll
            for (int d = 1; d < 16; d <<= 1) ps += __shfl_xor(ps, d, 32);
            rsum[i] = rsum[i] * alpha + ps;
            rmax[i] = nm;
#pragma unroll
            for (int t = 0; t < 5; ++t) oacc[t][i] *= alpha;
            p0[i] = e0; p1[i] = e1;
        }
#pragma unroll
        for (int i = 0; i < 8; ++i) {
            ptile[warp][i + half8][col]      = f2bf(p0[i]);
            ptile[warp][i + half8][col + 16] = f2bf(p1[i]);
        }
        __syncthreads();
        v16bf ap = wmma_load_a(&ptile[warp][0][0], 32, 0, 0, lane);
#pragma unroll
        for (int t = 0; t < 5; ++t) {
            v16bf bv = wmma_load_b(Vb, Mctx, t * 16, m0, lane);
            oacc[t] = wmma_bf16(ap, bv, oacc[t]);
        }
        __syncthreads();
    }

#pragma unroll
    for (int t = 0; t < 5; ++t) {
        int d = t * 16 + col;
        if (d >= DHEAD) continue;
#pragma unroll
        for (int i = 0; i < 8; ++i) {
            float denom = rsum[i];
            float v = (denom > 0.f) ? oacc[t][i] / denom : 0.f;
            O[((size_t)b * NLAT + nt * 16 + i + half8) * INNER_PAD + h * DHEAD + d] = f2bf(v);
        }
    }
}

// ---------------------------------------------------------------------------
// Host orchestration
// ---------------------------------------------------------------------------
extern "C" void kernel_launch(void* const* d_in, const int* in_sizes, int n_in,
                              void* d_out, int out_size, void* d_ws, size_t ws_size,
                              hipStream_t stream) {
    if (n_in < 65) return;
    const float* inp      = (const float*)d_in[0];
    const float* b_logits = (const float*)d_in[1];
    const float* latents  = (const float*)d_in[2];
    const float* pos_emb  = (const float*)d_in[63];
    const float* w_logits = (const float*)d_in[64];

    // workspace carve-up
    char* base = (char*)d_ws;
    size_t off = 0;
    auto take = [&](size_t bytes) -> char* {
        char* p = base + off;
        off = (off + bytes + 255) & ~(size_t)255;
        return p;
    };
    int*    order = (int*)   take((size_t)BQ * MM * 4);
    float*  maskf = (float*) take((size_t)BQ * MM * 4);
    bf16_t* data  = (bf16_t*)take((size_t)BQ * MM * DIN_PAD * 2);
    float*  zf    = (float*) take((size_t)BQ * NLAT * DLAT * 4);
    bf16_t* zb    = (bf16_t*)take((size_t)BQ * NLAT * DLAT * 2);
    bf16_t* qbuf  = (bf16_t*)take((size_t)BQ * HEADS * NLAT * DPAD * 2);
    bf16_t* Kc    = (bf16_t*)take((size_t)BQ * HEADS * MM * DPAD * 2);
    bf16_t* Vtc   = (bf16_t*)take((size_t)BQ * HEADS * 80 * MM * 2);
    bf16_t* Kl    = (bf16_t*)take((size_t)BQ * HEADS * NLAT * DPAD * 2);
    bf16_t* Vtl   = (bf16_t*)take((size_t)BQ * HEADS * 80 * NLAT * 2);
    bf16_t* obuf  = (bf16_t*)take((size_t)BQ * NLAT * INNER_PAD * 2);
    bf16_t* hbuf  = (bf16_t*)take((size_t)BQ * NLAT * 2048 * 2);

    bf16_t *wqT_ca[3], *wkvT_ca[3], *woT_ca[3], *w1T_c[3], *w2T_c[3];
    bf16_t *wqT_la[3], *wkvT_la[3], *woT_la[3], *w1T_l[3], *w2T_l[3];
    for (int L = 0; L < 3; ++L) {
        wqT_ca[L]  = (bf16_t*)take((size_t)INNER_PAD * DLAT * 2);
        wkvT_ca[L] = (bf16_t*)take((size_t)1200 * DIN_PAD * 2);
        woT_ca[L]  = (bf16_t*)take((size_t)DLAT * INNER_PAD * 2);
        w1T_c[L]   = (bf16_t*)take((size_t)2048 * DLAT * 2);
        w2T_c[L]   = (bf16_t*)take((size_t)DLAT * 2048 * 2);
        wqT_la[L]  = (bf16_t*)take((size_t)INNER_PAD * DLAT * 2);
        wkvT_la[L] = (bf16_t*)take((size_t)1200 * DLAT * 2);
        woT_la[L]  = (bf16_t*)take((size_t)DLAT * INNER_PAD * 2);
        w1T_l[L]   = (bf16_t*)take((size_t)2048 * DLAT * 2);
        w2T_l[L]   = (bf16_t*)take((size_t)DLAT * 2048 * 2);
    }
    bf16_t* wlogT = (bf16_t*)take((size_t)256 * DLAT * 2);
    (void)ws_size;

    // zero pad regions (head-dim pads stay zero across layers)
    hipMemsetAsync(qbuf, 0, (size_t)BQ * HEADS * NLAT * DPAD * 2, stream);
    hipMemsetAsync(Kc,   0, (size_t)BQ * HEADS * MM * DPAD * 2, stream);
    hipMemsetAsync(Vtc,  0, (size_t)BQ * HEADS * 80 * MM * 2, stream);
    hipMemsetAsync(Kl,   0, (size_t)BQ * HEADS * NLAT * DPAD * 2, stream);
    hipMemsetAsync(Vtl,  0, (size_t)BQ * HEADS * 80 * NLAT * 2, stream);
    hipMemsetAsync(obuf, 0, (size_t)BQ * NLAT * INNER_PAD * 2, stream);

    // sparsify + encode
    k_sparsify<<<BQ, 1024, 0, stream>>>(inp, order, maskf);
    k_encode<<<(BQ * MM + 255) / 256, 256, 0, stream>>>(inp, order, maskf, data);

    // weight casts
    auto castW = [&](const void* src, int K, int N, int Kpad, int Npad, bf16_t* dst) {
        int total = Npad * Kpad;
        k_cast_wT<<<(total + 255) / 256, 256, 0, stream>>>((const float*)src, K, N, Kpad, Npad, dst);
    };
    for (int L = 0; L < 3; ++L) {
        int bidx = 3 + 20 * L;
        castW(d_in[bidx + 4], DLAT, 600, DLAT, INNER_PAD, wqT_ca[L]);       // ca.wq
        castW(d_in[bidx + 2], 390, 1200, DIN_PAD, 1200,   wkvT_ca[L]);      // ca.wkv
        castW(d_in[bidx + 3], 600, DLAT, INNER_PAD, DLAT, woT_ca[L]);       // ca.wo
        castW(d_in[bidx + 8], DLAT, 2048, DLAT, 2048,     w1T_c[L]);        // cff.w1
        castW(d_in[bidx + 9], 2048, DLAT, 2048, DLAT,     w2T_c[L]);        // cff.w2
        castW(d_in[bidx + 14], DLAT, 600, DLAT, INNER_PAD, wqT_la[L]);      // la.wq
        castW(d_in[bidx + 12], DLAT, 1200, DLAT, 1200,     wkvT_la[L]);     // la.wkv
        castW(d_in[bidx + 13], 600, DLAT, INNER_PAD, DLAT, woT_la[L]);      // la.wo
        castW(d_in[bidx + 18], DLAT, 2048, DLAT, 2048,     w1T_l[L]);       // lff.w1
        castW(d_in[bidx + 19], 2048, DLAT, 2048, DLAT,     w2T_l[L]);       // lff.w2
    }
    castW(w_logits, DLAT, 256, DLAT, 256, wlogT);

    // latents init + single cast (mode-2 epilogues keep zb in sync afterwards)
    const int nz = BQ * NLAT * DLAT;
    k_init_z<<<(nz + 255) / 256, 256, 0, stream>>>(latents, pos_emb, zf);
    k_f32_to_bf16<<<(nz + 255) / 256, 256, 0, stream>>>(zf, zb, nz);

    auto qkv_grid  = [](int rows, int ncol) { return dim3((unsigned)((ncol + 63) / 64), (unsigned)(rows / 128)); };

    for (int L = 0; L < 3; ++L) {
        int bidx = 3 + 20 * L;
        const float* ca_bo    = (const float*)d_in[bidx + 0];
        const float* ca_scale = (const float*)d_in[bidx + 1];
        const float* cff_b1   = (const float*)d_in[bidx + 5];
        const float* cff_b2   = (const float*)d_in[bidx + 6];
        const float* cff_sc   = (const float*)d_in[bidx + 7];
        const float* la_bo    = (const float*)d_in[bidx + 10];
        const float* la_scale = (const float*)d_in[bidx + 11];
        const float* lff_b1   = (const float*)d_in[bidx + 15];
        const float* lff_b2   = (const float*)d_in[bidx + 16];
        const float* lff_sc   = (const float*)d_in[bidx + 17];

        // ---- cross attention ----
        k_gemm_qkv<<<qkv_grid(BQ * NLAT, INNER_PAD), 256, 0, stream>>>(
            zb, DLAT, wqT_ca[L], DLAT, INNER_PAD, DLAT, NLAT,
            nullptr, nullptr, qbuf, 0, 0);
        k_gemm_qkv<<<qkv_grid(BQ * MM, 1200), 256, 0, stream>>>(
            data, DIN_PAD, wkvT_ca[L], DIN_PAD, 1200, DIN_PAD, MM,
            Kc, Vtc, nullptr, MM, 1);
        k_attn<<<16, 256, 0, stream>>>(qbuf, Kc, Vtc, maskf, MM, obuf);
        k_gemm_bias<<<qkv_grid(BQ * NLAT, DLAT), 256, 0, stream>>>(
            obuf, INNER_PAD, woT_ca[L], INNER_PAD, DLAT, INNER_PAD,
            ca_bo, nullptr, nullptr, ca_scale, zf, zb, 2);

        // ---- cross FF ----
        k_gemm_bias<<<qkv_grid(BQ * NLAT, 2048), 256, 0, stream>>>(
            zb, DLAT, w1T_c[L], DLAT, 2048, DLAT,
            cff_b1, nullptr, hbuf, nullptr, nullptr, nullptr, 1);
        k_gemm_bias<<<qkv_grid(BQ * NLAT, DLAT), 256, 0, stream>>>(
            hbuf, 2048, w2T_c[L], 2048, DLAT, 2048,
            cff_b2, nullptr, nullptr, cff_sc, zf, zb, 2);

        // ---- latent self-attention ----
        k_gemm_qkv<<<qkv_grid(BQ * NLAT, INNER_PAD), 256, 0, stream>>>(
            zb, DLAT, wqT_la[L], DLAT, INNER_PAD, DLAT, NLAT,
            nullptr, nullptr, qbuf, 0, 0);
        k_gemm_qkv<<<qkv_grid(BQ * NLAT, 1200), 256, 0, stream>>>(
            zb, DLAT, wkvT_la[L], DLAT, 1200, DLAT, NLAT,
            Kl, Vtl, nullptr, NLAT, 1);
        k_attn<<<16, 256, 0, stream>>>(qbuf, Kl, Vtl, nullptr, NLAT, obuf);
        k_gemm_bias<<<qkv_grid(BQ * NLAT, DLAT), 256, 0, stream>>>(
            obuf, INNER_PAD, woT_la[L], INNER_PAD, DLAT, INNER_PAD,
            la_bo, nullptr, nullptr, la_scale, zf, zb, 2);

        // ---- latent FF ----
        k_gemm_bias<<<qkv_grid(BQ * NLAT, 2048), 256, 0, stream>>>(
            zb, DLAT, w1T_l[L], DLAT, 2048, DLAT,
            lff_b1, nullptr, hbuf, nullptr, nullptr, nullptr, 1);
        k_gemm_bias<<<qkv_grid(BQ * NLAT, DLAT), 256, 0, stream>>>(
            hbuf, 2048, w2T_l[L], 2048, DLAT, 2048,
            lff_b2, nullptr, nullptr, lff_sc, zf, zb, 2);
    }

    // final logits
    k_gemm_bias<<<qkv_grid(BQ * NLAT, 256), 256, 0, stream>>>(
        zb, DLAT, wlogT, DLAT, 256, DLAT,
        b_logits, (float*)d_out, nullptr, nullptr, nullptr, nullptr, 0);
}
